// NT_Xent_31018253811862
// MI455X (gfx1250) — matrix operand order repaired and checked by reading, not compiled, not run on hardware
//
#include <hip/hip_runtime.h>
#include <hip/hip_bf16.h>

// ---------------------------------------------------------------------------
// NT-Xent (SimCLR) loss, fused:
//   loss = (1/N) * sum_r [ log(sum_{c!=r} e^{s_rc}) - s_{r,partner(r)} ]
// with s = (zn @ zn^T) / TEMP, zn = row-L2-normalized concat(z_i, z_j).
// N = 8192, D = 128, TEMP = 0.5, partner(r) = r ^ 4096.
//
// Hot loop is fully uniform: 4x v_wmma_f32_16x16x32_f16 + 16-col exp-sum per
// tile, software-pipelined. The diagonal tile (ct == wave) and the partner
// tile (ct == wave ^ 256) are recomputed once afterwards for exact fix-up.
// ---------------------------------------------------------------------------

#define BATCH_B   4096
#define NROWS     8192          // N = 2*B
#define DIM       128
#define INV_TEMP  2.0f          // 1 / 0.5
#define INV_N     (1.0f / 8192.0f)
#define COL_TILES (NROWS / 16)  // 512

typedef _Float16 v4h  __attribute__((ext_vector_type(4)));
typedef _Float16 v8h  __attribute__((ext_vector_type(8)));
typedef _Float16 v16h __attribute__((ext_vector_type(16)));
typedef float    v8f  __attribute__((ext_vector_type(8)));

// ---------------------------------------------------------------------------
// Kernel 1: L2-normalize each row of concat(z_i, z_j) -> f16 matrix in ws.
// One wave32 per row; each lane owns 4 contiguous floats.
// ---------------------------------------------------------------------------
__global__ void ntxent_normalize(const float* __restrict__ zi,
                                 const float* __restrict__ zj,
                                 _Float16* __restrict__ znh) {
    const int row  = (blockIdx.x * blockDim.x + threadIdx.x) >> 5;  // exact: 8192
    const int lane = threadIdx.x & 31;

    const float* src = (row < BATCH_B) ? (zi + (size_t)row * DIM)
                                       : (zj + (size_t)(row - BATCH_B) * DIM);
    const float4 v = *(const float4*)(src + lane * 4);

    float ss = v.x * v.x + v.y * v.y + v.z * v.z + v.w * v.w;
#pragma unroll
    for (int mask = 16; mask >= 1; mask >>= 1)
        ss += __shfl_xor(ss, mask, 32);

    const float r = 1.0f / sqrtf(ss);

    v4h h;
    h[0] = (_Float16)(v.x * r);
    h[1] = (_Float16)(v.y * r);
    h[2] = (_Float16)(v.z * r);
    h[3] = (_Float16)(v.w * r);
    *(v4h*)(znh + (size_t)row * DIM + lane * 4) = h;   // global_store_b64
}

// Load one 16x32 f16 A-fragment chunk for this lane.
// Layout: lane<16 holds k = {0..15} of its row m=lane; lane>=16 holds
// k = {16..31}. `p` already includes the (hi ? 16 : 0) k-offset, so the two
// halves are p[0..7] and p[8..15].
__device__ __forceinline__ v16h load_a_frag(const _Float16* __restrict__ p) {
    const v8h lo = *(const v8h*)(p);        // global_load_b128
    const v8h hi = *(const v8h*)(p + 8);    // global_load_b128
    v16h r;
#pragma unroll
    for (int i = 0; i < 8; ++i) { r[i] = lo[i]; r[i + 8] = hi[i]; }
    return r;
}

// One 16x16 sim tile: 4 K-steps of v_wmma_f32_16x16x32_f16.
// `brow` = this lane's B pointer (column row base + k-half offset).
__device__ __forceinline__ v8f tile_mm(const v16h afr[4],
                                       const _Float16* __restrict__ brow) {
    v8f c = {};
#pragma unroll
    for (int ks = 0; ks < 4; ++ks) {
        const v16h b = *(const v16h*)(brow + 32 * ks);
        c = __builtin_amdgcn_wmma_f32_16x16x32_f16(
                false, afr[ks], false, b, (short)0, c, false, false);
    }
    return c;
}

// Uniform exp-sum of one tile's 8 C elements into per-row accumulators.
__device__ __forceinline__ void accum_tile(const v8f& c, float acc[8]) {
#pragma unroll
    for (int v = 0; v < 8; ++v)
        acc[v] += __expf(c[v] * INV_TEMP);
}

// ---------------------------------------------------------------------------
// Kernel 2: one wave per 16-row block; sweep all 512 column tiles (uniform,
// software-pipelined), then fix up diagonal + partner tiles, reduce, atomic.
// Grid is exact (512 waves) so EXEC is all-ones everywhere (WMMA requirement).
// ---------------------------------------------------------------------------
__global__ void ntxent_main(const _Float16* __restrict__ znh,
                            float* __restrict__ out) {
    const int wave = blockIdx.x * (blockDim.x >> 5) + (threadIdx.x >> 5);
    const int lane = threadIdx.x & 31;
    const int m    = lane & 15;          // A row-in-tile / B,C column-in-tile
    const bool hif = lane >= 16;

    const int r0 = wave * 16;            // this wave's 16 rows

    // Preload the 4 K-step A fragments (rows r0..r0+15, K = 0..127).
    v16h afr[4];
    {
        const _Float16* arow = znh + (size_t)(r0 + m) * DIM + (hif ? 16 : 0);
#pragma unroll
        for (int ks = 0; ks < 4; ++ks)
            afr[ks] = load_a_frag(arow + 32 * ks);
    }

    // Per-lane: 8 row partial exp-sums. C layout: N = m, M = (hif?8:0)+v.
    float acc[8];
#pragma unroll
    for (int v = 0; v < 8; ++v) acc[v] = 0.0f;

    // B fragment base for this lane: zn[col][k], k-half chosen by lane half.
    const _Float16* bbase = znh + (size_t)m * DIM + (hif ? 16 : 0);

    // ---- uniform hot loop, 2-stage pipeline (exp work hides WMMA hazards) --
    v8f cprev = tile_mm(afr, bbase);                 // tile ct = 0
    const _Float16* bp = bbase + 16 * DIM;
    for (int ct = 1; ct < COL_TILES; ++ct, bp += 16 * DIM) {
        const v8f ccur = tile_mm(afr, bp);           // loads + 4 WMMA (tile ct)
        accum_tile(cprev, acc);                      // exps of tile ct-1
        cprev = ccur;
    }
    accum_tile(cprev, acc);

    // ---- fix-up: recompute the 2 special tiles (bit-exact re-execution) ----
    // Diagonal tile: colBase = r0.  Partner tile: colBase = r0 ^ 4096.
    // In both, the interesting element for row M = (hif?8:0)+v sits at
    // lane column n = (hif?8:0)+v, i.e. the lane where d1 == v.
    const v8f cd = tile_mm(afr, bbase + (size_t)r0 * DIM);
    const v8f cp = tile_mm(afr, bbase + (size_t)(r0 ^ BATCH_B) * DIM);
    const int d1 = m - (hif ? 8 : 0);

    float pos[8];
#pragma unroll
    for (int v = 0; v < 8; ++v) {
        pos[v] = 0.0f;
        if (d1 == v) {
            acc[v] -= __expf(cd[v] * INV_TEMP);      // exact cancel of diag
            pos[v]  = cp[v] * INV_TEMP;              // positive logit
        }
    }

    // Reduce the 16 column-lanes of each half-wave (width-16 xor tree).
#pragma unroll
    for (int v = 0; v < 8; ++v) {
#pragma unroll
        for (int mask = 8; mask >= 1; mask >>= 1) {
            acc[v] += __shfl_xor(acc[v], mask, 16);
            pos[v] += __shfl_xor(pos[v], mask, 16);
        }
    }

    if (m == 0) {                         // lanes 0 and 16: 8 rows each
        float lsum = 0.0f;
#pragma unroll
        for (int v = 0; v < 8; ++v)
            lsum += logf(acc[v]) - pos[v];
        atomicAdd(out, lsum * INV_N);
    }
}

// ---------------------------------------------------------------------------
extern "C" void kernel_launch(void* const* d_in, const int* in_sizes, int n_in,
                              void* d_out, int out_size, void* d_ws, size_t ws_size,
                              hipStream_t stream) {
    const float* zi = (const float*)d_in[0];   // [4096,128] f32
    const float* zj = (const float*)d_in[1];   // [4096,128] f32
    float* out      = (float*)d_out;           // scalar f32 loss
    _Float16* znh   = (_Float16*)d_ws;         // [8192,128] f16 (2 MB)

    (void)in_sizes; (void)n_in; (void)out_size; (void)ws_size;

    hipMemsetAsync(out, 0, sizeof(float), stream);

    // 8192 rows, one wave each: 1024 blocks x 256 threads (8 waves/block).
    ntxent_normalize<<<NROWS / 8, 256, 0, stream>>>(zi, zj, znh);

    // 512 row-blocks, one wave each: 64 blocks x 256 threads (8 waves/block).
    ntxent_main<<<(NROWS / 16) / 8, 256, 0, stream>>>(znh, out);
}
